// Controller_44933947851380
// MI455X (gfx1250) — compile-verified, matrix operand order checked by reading, not run on hardware
//
#include <hip/hip_runtime.h>
#include <hip/hip_bf16.h>

// ---------------------------------------------------------------------------
// ENAS controller on MI455X (gfx1250).
// 96 sequential LSTM steps, 2 layers, hidden 1024.
//   * fp32 weights -> bf16 once per launch, pre-swizzled into WMMA fragment
//     order: per (layer,tile,kchunk) a 1KB block = 32 lanes x 32B, SoA
//     (h[0] halves then h[1] halves) -> every A load is fully coalesced.
//     33.5 MB of bf16 weights fit in the 192 MB L2 -> per-step streaming is
//     L2-resident, not HBM-bound.
//   * gates = Wcat @ [x;h] via V_WMMA_F32_16X16X32_BF16, f32 accumulation.
//   * x-vector staged to LDS via GLOBAL_LOAD_ASYNC_TO_LDS_B128 (ASYNCcnt
//     path) when the builtin exists, else plain copy.
//   * one persistent kernel, grid-wide barriers via an atomic counter.
// ---------------------------------------------------------------------------

typedef __attribute__((ext_vector_type(16))) __bf16 v16bf;
typedef __attribute__((ext_vector_type(8)))  __bf16 v8bf;
typedef __attribute__((ext_vector_type(8)))  float  v8f;
typedef __attribute__((ext_vector_type(4)))  int    v4i;

union BFrag { v16bf v; v8bf h[2]; };

#define L_SIZE      1024
#define GATE_ROWS   4096              // 4*L
#define KDIM        2048              // [x ; h]
#define NLAYERS     2
#define NSTEPS      96
#define NBLK        64
#define NTHR        256
#define WAVES_PER_BLK 8
#define NTILES      256               // GATE_ROWS / 16
#define KCHUNKS     64                // KDIM / 32
#define FRAG_HALVES 512               // 32 lanes * 16 halves = 1KB per fragment

// workspace layout (bytes)
#define WCAT_OFF    0ull                                   // bf16, fragment-swizzled
#define WCAT_BYTES  (2ull*GATE_ROWS*KDIM*2ull)             // 33,554,432
#define BCOMB_OFF   (WCAT_OFF + WCAT_BYTES)                // f32 [2][4096]
#define BCOMB_BYTES (2ull*GATE_ROWS*4ull)
#define XCAT0_OFF   (BCOMB_OFF + BCOMB_BYTES)              // bf16 [2048] layer-0 input
#define XCAT1_OFF   (XCAT0_OFF + KDIM*2ull)                // bf16 [2048] layer-1 input
#define CSTATE_OFF  (XCAT1_OFF + KDIM*2ull)                // f32 [2][1024]
#define GATES_OFF   (CSTATE_OFF + 2ull*L_SIZE*4ull)        // f32 [4096]
#define CNT_OFF     (GATES_OFF + GATE_ROWS*4ull)           // u32 barrier counter

// ------------------------- helpers -----------------------------------------

__device__ __forceinline__ float sigmoidf_(float x) { return 1.0f / (1.0f + __expf(-x)); }

__device__ __forceinline__ float u01_(unsigned step) {
    unsigned long long z = 0x9E3779B97F4A7C15ull * (unsigned long long)(step + 0x2A7B1uL);
    z ^= z >> 30; z *= 0xBF58476D1CE4E5B9ull;
    z ^= z >> 27; z *= 0x94D049BB133111EBull;
    z ^= z >> 31;
    return ((float)(unsigned)(z >> 40) + 0.5f) * (1.0f / 16777216.0f);
}

__device__ __forceinline__ void grid_barrier(unsigned* cnt, unsigned epoch) {
    __syncthreads();
    if (threadIdx.x == 0) {
        __threadfence();
        __hip_atomic_fetch_add(cnt, 1u, __ATOMIC_RELEASE, __HIP_MEMORY_SCOPE_AGENT);
        const unsigned goal = epoch * NBLK;
        while (__hip_atomic_load(cnt, __ATOMIC_ACQUIRE, __HIP_MEMORY_SCOPE_AGENT) < goal)
            __builtin_amdgcn_s_sleep(1);
        __threadfence();
    }
    __syncthreads();
}

// Stage 4KB x-vector into LDS. Prefer the gfx1250 async-to-LDS path.
// Builtin signature (from hipcc diagnostic): param0 is v4i addrspace(1)*.
__device__ __forceinline__ void stage_x(const __bf16* __restrict__ src,
                                        __bf16* xs, unsigned tid) {
#if __has_builtin(__builtin_amdgcn_global_load_async_to_lds_b128)
    typedef __attribute__((address_space(1))) v4i* gv4p;
    typedef __attribute__((address_space(3))) v4i* lv4p;
    __builtin_amdgcn_global_load_async_to_lds_b128(
        (gv4p)(src + tid * 8u), (lv4p)(xs + tid * 8u), 0, 0);
    asm volatile("s_wait_asynccnt 0x0" ::: "memory");
#else
    ((v8bf*)xs)[tid] = ((const v8bf*)src)[tid];
#endif
}

// ------------------------- setup kernels ------------------------------------

// Build fragment-swizzled bf16 weights:
//   frag = ((layer*256 + tile)*64 + kidx), 512 halves each.
//   within-frag: [hi(0/1)][lane(0..31)][j8(0..7)] ->
//     element W[tile*16 + (lane&15)][kidx*32 + hi*16 + (lane>>4)*8 + j8]
// so in the matvec, lane reads frag + lane*8 (h0) and frag + 256 + lane*8 (h1):
// both loads fully coalesced (contiguous 512B per wave per load).
__global__ void convert_weights_kernel(const float* __restrict__ w_ih,
                                       const float* __restrict__ w_hh,
                                       const float* __restrict__ b_ih,
                                       const float* __restrict__ b_hh,
                                       __bf16* __restrict__ wcat,
                                       float*  __restrict__ bcomb) {
    const unsigned total = 2u * GATE_ROWS * KDIM;          // 16,777,216
    for (unsigned idx = blockIdx.x * blockDim.x + threadIdx.x; idx < total;
         idx += gridDim.x * blockDim.x) {
        const unsigned within = idx & 511u;
        const unsigned j8     = within & 7u;
        const unsigned lane   = (within >> 3) & 31u;
        const unsigned hi     = within >> 8;               // 0 or 1
        const unsigned kidx   = (idx >> 9)  & 63u;
        const unsigned tile   = (idx >> 15) & 255u;
        const unsigned layer  = idx >> 23;
        const unsigned m   = lane & 15u;
        const unsigned g   = lane >> 4;
        const unsigned row = tile * 16u + m;
        const unsigned k   = kidx * 32u + hi * 16u + g * 8u + j8;
        float v;
        if (k < 1024u) v = w_ih[((size_t)layer * GATE_ROWS + row) * L_SIZE + k];
        else           v = w_hh[((size_t)layer * GATE_ROWS + row) * L_SIZE + (k - 1024u)];
        wcat[idx] = (__bf16)v;
        if (idx < 2u * GATE_ROWS)                          // fused bias combine
            bcomb[idx] = b_ih[idx] + b_hh[idx];
    }
}

__global__ void init_state_kernel(const float* __restrict__ g_emb,
                                  __bf16* __restrict__ xcat0,
                                  __bf16* __restrict__ xcat1,
                                  float*  __restrict__ cstate,
                                  unsigned* __restrict__ cnt) {
    for (unsigned t = blockIdx.x * blockDim.x + threadIdx.x; t < KDIM;
         t += gridDim.x * blockDim.x) {
        xcat0[t] = (t < L_SIZE) ? (__bf16)g_emb[t] : (__bf16)0.0f;
        xcat1[t] = (__bf16)0.0f;
        cstate[t] = 0.0f;                                  // covers [2][1024]
    }
    if (blockIdx.x == 0 && threadIdx.x == 0) *cnt = 0u;
}

// ------------------------- persistent controller ----------------------------

__global__ void __launch_bounds__(NTHR, 1)
controller_kernel(const __bf16* __restrict__ wcat,
                  const float*  __restrict__ bcomb,
                  __bf16* __restrict__ xcat0,
                  __bf16* __restrict__ xcat1,
                  float*  __restrict__ cstate,
                  float*  __restrict__ gates,
                  unsigned* __restrict__ cnt,
                  const float* __restrict__ wsoft,         // [2,1024] f32
                  const float* __restrict__ wemb,          // [8,1024] f32
                  float* __restrict__ out) {               // [96 + 2]
    __shared__ __bf16 xs[KDIM];                            // 4 KB shared input vector
    __shared__ float  cred[4][32][8];                      // 4 KB wave-pair reduction

    const unsigned tid   = threadIdx.x;
    const unsigned lane  = tid & 31u;
    const unsigned waveL = tid >> 5;                       // wave in block, 0..7
    const unsigned waveG = blockIdx.x * WAVES_PER_BLK + waveL;
    const unsigned tile  = waveG >> 1;                     // 0..255 (16 rows each)
    const unsigned khalf = waveG & 1u;                     // K split: [0,1024) / [1024,2048)
    const unsigned tileL = waveL >> 1;                     // tile index within block, 0..3
    const unsigned m     = lane & 15u;                     // row within tile
    const unsigned g     = lane >> 4;                      // half-wave group
    const unsigned gtid  = blockIdx.x * NTHR + tid;

    // B-fragment K pattern (CDNA5 CDNA5 ISA §7.12.2, 16-bit operands):
    // halves 0..7 -> K = g*8 + j ; halves 8..15 -> K = 16 + g*8 + j
    const unsigned kofsA = g * 8u;
    const unsigned kofsB = 16u + g * 8u;

    unsigned ep = 0u;
    float entSum = 0.0f, lpSum = 0.0f;                     // live in sampler wave regs

    for (unsigned step = 0; step < NSTEPS; ++step) {
        for (unsigned layer = 0; layer < NLAYERS; ++layer) {
            // ---- stage x-vector into LDS (2048 bf16 = 4 KB)
            stage_x((layer == 0) ? xcat0 : xcat1, xs, tid);
            __syncthreads();

            // ---- 16-row x 1024-K partial mat-vec per wave: 32 WMMAs,
            //      A fragments from swizzled weights, fully coalesced.
            const unsigned R  = tile * 16u;
            const unsigned KO = khalf * 1024u;
            const __bf16* frag0 =
                wcat + ((((size_t)layer * NTILES + tile) * KCHUNKS + khalf * 32u)
                        << 9) + lane * 8u;
            v8f acc = {};
            #pragma unroll 4
            for (unsigned i = 0; i < 32u; ++i) {
                BFrag a, b;
                a.h[0] = *(const v8bf*)(frag0 + i * FRAG_HALVES);
                a.h[1] = *(const v8bf*)(frag0 + i * FRAG_HALVES + 256u);
                const unsigned kb = KO + i * 32u;
                b.h[0] = *(const v8bf*)(xs + kb + kofsA);   // broadcast cols
                b.h[1] = *(const v8bf*)(xs + kb + kofsB);
                acc = __builtin_amdgcn_wmma_f32_16x16x32_bf16(
                          false, a.v, false, b.v, (short)0, acc, false, false);
            }

            // ---- combine K-halves through LDS, write 16 rows of gates
            if (khalf == 1u) {
                #pragma unroll
                for (int r = 0; r < 8; ++r) cred[tileL][lane][r] = acc[r];
            }
            __syncthreads();
            if (khalf == 0u) {
                #pragma unroll
                for (int r = 0; r < 8; ++r) acc[r] += cred[tileL][lane][r];
                if (m == 0u) {                              // lanes 0 and 16 (N==0)
                    #pragma unroll
                    for (int r = 0; r < 8; ++r)
                        gates[R + (unsigned)r + 8u * g] = acc[r];
                }
            }
            grid_barrier(cnt, ++ep);

            // ---- elementwise LSTM cell update (1024 lanes across grid)
            if (gtid < L_SIZE) {
                const unsigned j  = gtid;
                const float* bc = bcomb + (size_t)layer * GATE_ROWS;
                const float gi = gates[j]            + bc[j];
                const float gf = gates[1024u + j]    + bc[1024u + j];
                const float gg = gates[2048u + j]    + bc[2048u + j];
                const float go = gates[3072u + j]    + bc[3072u + j];
                float c = sigmoidf_(gf) * cstate[layer * L_SIZE + j] +
                          sigmoidf_(gi) * tanhf(gg);
                float h = sigmoidf_(go) * tanhf(c);
                cstate[layer * L_SIZE + j] = c;
                const __bf16 hb = (__bf16)h;
                if (layer == 0) {
                    xcat1[j] = hb;                          // next layer input
                    xcat0[L_SIZE + j] = hb;                 // h0 for next step
                } else {
                    xcat1[L_SIZE + j] = hb;                 // h1 for next step + softmax
                }
            }
            grid_barrier(cnt, ++ep);
        }

        // ---- softmax head + categorical sample (one wave)
        if (blockIdx.x == 0 && waveL == 0) {
            float a0 = 0.0f, a1 = 0.0f;
            for (unsigned k = lane; k < L_SIZE; k += 32u) {
                const float h = (float)xcat1[L_SIZE + k];
                a0 += wsoft[k] * h;
                a1 += wsoft[L_SIZE + k] * h;
            }
            #pragma unroll
            for (int off = 16; off > 0; off >>= 1) {
                a0 += __shfl_down(a0, off, 32);
                a1 += __shfl_down(a1, off, 32);
            }
            int sample = 0;
            if (lane == 0) {
                const float l0 = 2.5f * tanhf(a0 * 0.2f);
                const float l1 = 2.5f * tanhf(a1 * 0.2f);
                const float mx = fmaxf(l0, l1);
                const float z  = __logf(__expf(l0 - mx) + __expf(l1 - mx));
                const float lp0 = l0 - mx - z, lp1 = l1 - mx - z;
                const float p0 = __expf(lp0), p1 = __expf(lp1);
                sample = (u01_(step) < p0) ? 0 : 1;
                lpSum  += (sample == 0) ? lp0 : lp1;
                entSum += -(p0 * lp0 + p1 * lp1);
                out[step] = (float)sample;
            }
            sample = __shfl(sample, 0, 32);
            for (unsigned k = lane; k < L_SIZE; k += 32u)   // embedding feedback
                xcat0[k] = (__bf16)wemb[(size_t)sample * L_SIZE + k];
        }
        grid_barrier(cnt, ++ep);
    }

    if (blockIdx.x == 0 && tid == 0) {
        out[NSTEPS]     = entSum;                           // sum(entropies)
        out[NSTEPS + 1] = lpSum;                            // sum(log_probs)
    }
}

// ------------------------- host glue ----------------------------------------

extern "C" void kernel_launch(void* const* d_in, const int* in_sizes, int n_in,
                              void* d_out, int out_size, void* d_ws, size_t ws_size,
                              hipStream_t stream) {
    const float* g_emb = (const float*)d_in[0];   // [1,1024]
    const float* w_emb = (const float*)d_in[1];   // [8,1024]
    const float* wsoft = (const float*)d_in[2];   // [2,1024]
    const float* w_ih  = (const float*)d_in[3];   // [2,4096,1024]
    const float* w_hh  = (const float*)d_in[4];   // [2,4096,1024]
    const float* b_ih  = (const float*)d_in[5];   // [2,4096]
    const float* b_hh  = (const float*)d_in[6];   // [2,4096]

    char* ws = (char*)d_ws;
    __bf16*   wcat   = (__bf16*)(ws + WCAT_OFF);
    float*    bcomb  = (float*)(ws + BCOMB_OFF);
    __bf16*   xcat0  = (__bf16*)(ws + XCAT0_OFF);
    __bf16*   xcat1  = (__bf16*)(ws + XCAT1_OFF);
    float*    cstate = (float*)(ws + CSTATE_OFF);
    float*    gates  = (float*)(ws + GATES_OFF);
    unsigned* cnt    = (unsigned*)(ws + CNT_OFF);

    convert_weights_kernel<<<8192, 256, 0, stream>>>(w_ih, w_hh, b_ih, b_hh,
                                                     wcat, bcomb);
    init_state_kernel<<<8, 256, 0, stream>>>(g_emb, xcat0, xcat1, cstate, cnt);
    controller_kernel<<<NBLK, NTHR, 0, stream>>>(wcat, bcomb, xcat0, xcat1,
                                                 cstate, gates, cnt,
                                                 wsoft, w_emb, (float*)d_out);
}
